// LSTM_83167746720299
// MI455X (gfx1250) — matrix operand order repaired and checked by reading, not compiled, not run on hardware
//
#include <hip/hip_runtime.h>

// ---------------------------------------------------------------------------
// LSTM (SEQ=40, H=4096, input_size=1) + dense(128) head for MI455X / gfx1250.
//
// Roofline: fp32 w_hh (256MB) exceeds the 192MB L2, so fp32 steps would
// stream HBM (40 x 256MB / 23.3TB/s ~ 440us). bf16 w_hh (128MB) fits L2:
// one-time convert+reorder, then all 40 sequential matvec steps are
// L2-resident. Matvec runs on v_wmma_f32_16x16x32_bf16 with the h-vector
// broadcast across all 16 B columns (A-side bytes are the bound; wasted
// B-side FLOPs are free). Row reorder (4n+g) makes each 16-row WMMA tile
// hold 4 complete hidden units, fusing gate nonlinearities into the matvec.
// ---------------------------------------------------------------------------

#define SEQ 40
#define H   4096
#define NG  (4 * H)       // 16384 gate rows
#define OUTN 128

typedef __bf16 v4bf  __attribute__((ext_vector_type(4)));
typedef __bf16 v8bf  __attribute__((ext_vector_type(8)));
typedef __bf16 v16bf __attribute__((ext_vector_type(16)));
typedef float  v8f   __attribute__((ext_vector_type(8)));

static __device__ __forceinline__ float sigf(float x) {
    return 1.0f / (1.0f + __expf(-x));
}

// ---------------------------------------------------------------------------
// Prep 1: convert + reorder w_hh (fp32, row_old = g*H + n, row-major, K=4096)
// into Wr (bf16, row_new = 4*n + g). Each thread handles 4 consecutive k of
// one row: fully coalesced float4 reads, 8B bf16 writes.
// grid: 65536 x 256  (16384 rows * 1024 quads)
// ---------------------------------------------------------------------------
__global__ void lstm_prep_w(const float* __restrict__ w_hh,
                            __bf16* __restrict__ Wr) {
    unsigned int flat = blockIdx.x * blockDim.x + threadIdx.x;   // 0 .. 16M-1
    unsigned int row_old = flat >> 10;           // 0..16383
    unsigned int k4      = (flat & 1023u) << 2;  // 0..4092
    float4 v = *(const float4*)(w_hh + (size_t)row_old * H + k4);
    unsigned int n = row_old & (H - 1);          // unit index
    unsigned int g = row_old >> 12;              // gate index 0..3 (i,f,g,o)
    unsigned int row_new = (n << 2) | g;
    v4bf o;
    o[0] = (__bf16)v.x; o[1] = (__bf16)v.y; o[2] = (__bf16)v.z; o[3] = (__bf16)v.w;
    *(v4bf*)(Wr + (size_t)row_new * H + k4) = o;
}

// ---------------------------------------------------------------------------
// Prep 2: reorder fused bias + w_ih column; zero c and h ping-pong buffers.
// grid: 64 x 256  (16384 threads)
// ---------------------------------------------------------------------------
__global__ void lstm_prep_small(const float* __restrict__ w_ih,
                                const float* __restrict__ b_ih,
                                const float* __restrict__ b_hh,
                                float* __restrict__ br,
                                float* __restrict__ wihr,
                                float* __restrict__ c,
                                __bf16* __restrict__ hA,
                                __bf16* __restrict__ hB) {
    unsigned int j = blockIdx.x * blockDim.x + threadIdx.x;      // 0..16383
    unsigned int g = j & 3u;
    unsigned int n = j >> 2;
    unsigned int row_old = g * H + n;
    br[j]   = b_ih[row_old] + b_hh[row_old];
    wihr[j] = w_ih[row_old];              // w_ih is (4H, 1)
    if (j < H) {
        c[j]  = 0.0f;
        hA[j] = (__bf16)0.0f;
        hB[j] = (__bf16)0.0f;
    }
}

// ---------------------------------------------------------------------------
// One LSTM timestep. Block = 128 threads = 4 waves, owns one 16-row tile
// (= 4 hidden units). Wave w accumulates K range [w*1024, w*1024+1024) with
// 32 WMMA 16x16x32 bf16 ops (compile-time trip count -> scalar loop, no
// per-iteration EXEC manipulation); h staged in LDS; gate partials reduced
// via LDS float atomics; 4 threads apply nonlinearities and update c / h.
// grid: 1024 x 128
// ---------------------------------------------------------------------------
__global__ void lstm_step(const __bf16* __restrict__ Wr,
                          const float* __restrict__ br,
                          const float* __restrict__ wihr,
                          const float* __restrict__ x, int t,
                          const __bf16* __restrict__ h_in,
                          __bf16* __restrict__ h_out,
                          float* __restrict__ c) {
    __shared__ __bf16 hS[H];
    __shared__ float  gateS[16];

    const unsigned int tid = threadIdx.x;

    // Stage h (8KB bf16) into LDS: each thread copies 32 elements (2 x 32B).
    {
        unsigned int base = tid * 32u;
        *(v16bf*)(hS + base)      = *(const v16bf*)(h_in + base);
        *(v16bf*)(hS + base + 16) = *(const v16bf*)(h_in + base + 16);
    }
    if (tid < 16) gateS[tid] = 0.0f;
    __syncthreads();

    const unsigned int lane = tid & 31u;
    const unsigned int wave = tid >> 5;
    const unsigned int m    = lane & 15u;   // A-matrix row within tile
    const unsigned int hi   = lane >> 4;    // lane half selects K sub-ranges

    // A fragment (16-bit 16x32, contiguous K order matching B): lane<16 holds K 0..15,
    // lane>=16 holds K 16..31 of row m.
    const unsigned int kStart = wave * 1024u;
    const __bf16* pa = Wr + ((size_t)blockIdx.x * 16 + m) * H + kStart + hi * 16u;
    const __bf16* pb = hS + kStart + hi * 16u;

    v8f acc = {};
#pragma unroll 4
    for (int i = 0; i < 32; ++i) {          // constant trip count: scalar loop
        v8bf alo = *(const v8bf*)(pa);
        v8bf ahi = *(const v8bf*)(pa + 8);
        v16bf a = __builtin_shufflevector(alo, ahi,
                    0, 1, 2, 3, 4, 5, 6, 7, 8, 9, 10, 11, 12, 13, 14, 15);
        // B fragment: broadcast h chunk across all 16 columns.
        // Lane half 0 holds K 0..15, half 1 holds K 16..31 of the chunk.
        v16bf b = *(const v16bf*)(pb);
        acc = __builtin_amdgcn_wmma_f32_16x16x32_bf16(
                  false, a, false, b, (short)0, acc, false, false);
        pa += 32;
        pb += 32;
    }

    // D layout: all columns identical; lane 0 holds rows 0-7 in acc[0..7],
    // lane 16 holds rows 8-15. Reduce the 4 K-split waves via LDS atomics.
    if (m == 0) {
#pragma unroll
        for (int r = 0; r < 8; ++r)
            atomicAdd(&gateS[hi * 8u + r], acc[r]);
    }
    __syncthreads();

    if (tid < 4) {
        const float xt = x[t];
        const unsigned int rb = blockIdx.x * 16u + tid * 4u;  // reordered rows
        float gi = gateS[tid * 4 + 0] + br[rb + 0] + wihr[rb + 0] * xt;
        float gf = gateS[tid * 4 + 1] + br[rb + 1] + wihr[rb + 1] * xt;
        float gg = gateS[tid * 4 + 2] + br[rb + 2] + wihr[rb + 2] * xt;
        float go = gateS[tid * 4 + 3] + br[rb + 3] + wihr[rb + 3] * xt;
        const unsigned int unit = blockIdx.x * 4u + tid;
        float cn = sigf(gf) * c[unit] + sigf(gi) * tanhf(gg);
        float hn = sigf(go) * tanhf(cn);
        c[unit]     = cn;
        h_out[unit] = (__bf16)hn;
    }
}

// ---------------------------------------------------------------------------
// Dense head: out[o] = dense_w[o,:] @ h_last + dense_b[o].
// grid: 128 blocks x 256 threads, tree-reduce in LDS.
// ---------------------------------------------------------------------------
__global__ void lstm_dense(const float* __restrict__ W,
                           const float* __restrict__ bias,
                           const __bf16* __restrict__ h,
                           float* __restrict__ out) {
    __shared__ float red[256];
    const float* row = W + (size_t)blockIdx.x * H;
    float s = 0.0f;
    for (unsigned int k = threadIdx.x; k < H; k += 256u)
        s += row[k] * (float)h[k];
    red[threadIdx.x] = s;
    __syncthreads();
    for (unsigned int off = 128; off > 0; off >>= 1) {
        if (threadIdx.x < off) red[threadIdx.x] += red[threadIdx.x + off];
        __syncthreads();
    }
    if (threadIdx.x == 0) out[blockIdx.x] = red[0] + bias[blockIdx.x];
}

// ---------------------------------------------------------------------------
extern "C" void kernel_launch(void* const* d_in, const int* in_sizes, int n_in,
                              void* d_out, int out_size, void* d_ws, size_t ws_size,
                              hipStream_t stream) {
    const float* x       = (const float*)d_in[0];
    const float* w_ih    = (const float*)d_in[1];
    const float* w_hh    = (const float*)d_in[2];
    const float* b_ih    = (const float*)d_in[3];
    const float* b_hh    = (const float*)d_in[4];
    const float* dense_w = (const float*)d_in[5];
    const float* dense_b = (const float*)d_in[6];
    float* out = (float*)d_out;

    // Workspace carving (~128.2 MB total, all chunks >=4KB aligned):
    char* p = (char*)d_ws;
    __bf16* Wr   = (__bf16*)p;  p += (size_t)NG * H * sizeof(__bf16);  // 128 MB
    float*  br   = (float*)p;   p += (size_t)NG * sizeof(float);
    float*  wihr = (float*)p;   p += (size_t)NG * sizeof(float);
    float*  c    = (float*)p;   p += (size_t)H * sizeof(float);
    __bf16* hA   = (__bf16*)p;  p += (size_t)H * sizeof(__bf16);
    __bf16* hB   = (__bf16*)p;  p += (size_t)H * sizeof(__bf16);
    (void)ws_size; (void)in_sizes; (void)n_in; (void)out_size;

    // One-time (per call) weight convert/reorder + state init.
    lstm_prep_w<<<65536, 256, 0, stream>>>(w_hh, Wr);
    lstm_prep_small<<<64, 256, 0, stream>>>(w_ih, b_ih, b_hh, br, wihr, c, hA, hB);

    // 40 sequential timesteps; stream ordering is the recurrence barrier.
    for (int t = 0; t < SEQ; ++t) {
        const __bf16* hin = (t & 1) ? hB : hA;
        __bf16*       hout = (t & 1) ? hA : hB;
        lstm_step<<<NG / 16, 128, 0, stream>>>(Wr, br, wihr, x, t, hin, hout, c);
    }
    // t=39 (odd) wrote hA -> final hidden state.
    lstm_dense<<<OUTN, 256, 0, stream>>>(dense_w, dense_b, hA, out);
}
